// LeftSubNet_74182675137216
// MI455X (gfx1250) — matrix operand order repaired and checked by reading, not compiled, not run on hardware
//
#include <hip/hip_runtime.h>
#include <hip/hip_bf16.h>

// ---------------------------------------------------------------------------
// S4D network for MI455X (gfx1250, wave32).
// Chunked-scan SSM: phase1/phase3/Wo are f16 WMMA GEMMs with f32 accumulate.
// Fragment operands staged in LDS so every WMMA operand load is ds_load_b128.
// ---------------------------------------------------------------------------

typedef __attribute__((ext_vector_type(16))) _Float16 v16h;
typedef __attribute__((ext_vector_type(8)))  float    v8f;

constexpr int Bn = 32, Hn = 64, Ln = 4096, Nn = 64, Tn = 64, NCn = 64; // NC = L/T

// ---- workspace layout (bytes) ----
constexpr size_t SZ_ACT  = (size_t)Bn*Hn*Ln*4;          // 33.55 MB
constexpr size_t OFF_BUFA = 0;
constexpr size_t OFF_BUFB = OFF_BUFA + SZ_ACT;
constexpr size_t SZ_S    = (size_t)NCn*Hn*128*Bn*4;      // 67.1 MB (chunk states, re/im stacked)
constexpr size_t OFF_S   = OFF_BUFB + SZ_ACT;
constexpr size_t SZ_G    = (size_t)Bn*Hn*Ln*2;           // 16.8 MB (gelu output, f16)
constexpr size_t OFF_G   = OFF_S + SZ_S;
constexpr size_t SZ_A1   = (size_t)Hn*128*Tn*2;          // phase1 coeff matrices
constexpr size_t OFF_A1  = OFF_G + SZ_G;
constexpr size_t SZ_A3   = (size_t)Hn*Tn*192*2;          // phase3 [Toeplitz | state-inject]
constexpr size_t OFF_A3  = OFF_A1 + SZ_A1;
constexpr size_t OFF_ZT  = OFF_A3 + SZ_A3;               // z^T per (h,n), float2
constexpr size_t SZ_ZT   = (size_t)Hn*Nn*8;
constexpr size_t OFF_ZCB = OFF_ZT + SZ_ZT;               // z, Cb per (h,n), float4
constexpr size_t SZ_ZCB  = (size_t)Hn*Nn*16;
constexpr size_t OFF_STP = OFF_ZCB + SZ_ZCB;             // BN partials [2][64][32]
constexpr size_t OFF_STF = OFF_STP + 4096*4;             // BN final [2][64]
constexpr size_t WS_NEED = OFF_STF + 512;

__device__ __forceinline__ float gelu_tanh(float x) {
  const float k = 0.7978845608028654f;
  return 0.5f*x*(1.f + tanhf(k*(x + 0.044715f*x*x*x)));
}

// ---- WMMA fragment helpers (ISA 7.12.2 layouts, wave32) ----
// Both halves of every fragment are 16B-contiguous in LDS -> ds_load_b128 pairs.
__device__ __forceinline__ v16h frag_pair(const _Float16* p0, const _Float16* p1) {
  union { uint4 u[2]; v16h v; } r;
  r.u[0] = *(const uint4*)p0;
  r.u[1] = *(const uint4*)p1;
  return r.v;
}
// A: 16x32, row-major LDS with padded stride lda (multiple of 8, 16B-aligned rows)
__device__ __forceinline__ v16h frag_a(const _Float16* A, int lda, int row0, int k0, int lane) {
  int m = row0 + (lane & 15), kh = (lane >> 4) * 8;
  const _Float16* base = A + m*lda + k0 + kh;
  return frag_pair(base, base + 16);
}
// B: 32x16, COLUMN-major LDS (Bc[n*ldk + k]), padded stride ldk
__device__ __forceinline__ v16h frag_b(const _Float16* Bc, int ldk, int k0, int n0, int lane) {
  int n = n0 + (lane & 15), kh = (lane >> 4) * 8;
  const _Float16* base = Bc + n*ldk + k0 + kh;
  return frag_pair(base, base + 16);
}

// ===========================================================================
// S4 block kernels
// ===========================================================================

// z = exp(dt*A); Cb = C*(z-1)/A   (one thread per (h,n))
__global__ void s4_prep(const float* __restrict__ log_dt, const float* __restrict__ A_re_log,
                        const float* __restrict__ A_im,  const float* __restrict__ C_re,
                        const float* __restrict__ C_im,  float4* __restrict__ zcb) {
  int idx = blockIdx.x*256 + threadIdx.x;     // H*N = 4096
  int h = idx >> 6;
  float dt  = expf(log_dt[h]);
  float are = -expf(A_re_log[idx]);
  float aim = A_im[idx];
  float dre = are*dt, dim_ = aim*dt;
  float ex  = expf(dre);
  float zre = ex*cosf(dim_), zim = ex*sinf(dim_);
  float wre = zre - 1.f, wim = zim;
  float den = are*are + aim*aim;
  float qre = (wre*are + wim*aim)/den;
  float qim = (wim*are - wre*aim)/den;
  float cre = C_re[idx], cim = C_im[idx];
  zcb[idx] = make_float4(zre, zim, cre*qre - cim*qim, cre*qim + cim*qre);
}

// Build per-h chunk matrices from powers z^0..z^T (one workgroup per h, 64 threads)
__global__ void s4_chunkmat(const float4* __restrict__ zcb, _Float16* __restrict__ A1,
                            _Float16* __restrict__ A3, float2* __restrict__ ZT) {
  int h = blockIdx.x, n = threadIdx.x;
  __shared__ float pr[64][65], pi[64][65];
  __shared__ float cbr[64], cbi[64], Ks[64];
  float4 p = zcb[h*64 + n];
  cbr[n] = p.z; cbi[n] = p.w;
  float re = 1.f, im = 0.f;
  for (int d = 0; d <= 64; ++d) {
    pr[n][d] = re; pi[n][d] = im;
    float nre = re*p.x - im*p.y, nim = re*p.y + im*p.x;
    re = nre; im = nim;
  }
  __syncthreads();
  { // SSM kernel K[d] = 2*Re(sum_n Cb_n z_n^d)
    int d = n; float s = 0.f;
    for (int k = 0; k < 64; ++k) s += 2.f*(cbr[k]*pr[k][d] - cbi[k]*pi[k][d]);
    Ks[d] = s;
  }
  __syncthreads();
  // phase1 A: rows n -> Re(z^{T-1-k}), rows 64+n -> Im(z^{T-1-k})
  _Float16* a1 = A1 + (size_t)h*128*64;
  for (int k = 0; k < 64; ++k) {
    a1[n*64 + k]      = (_Float16)pr[n][63-k];
    a1[(64+n)*64 + k] = (_Float16)pi[n][63-k];
  }
  // phase3 A: cols 0..63 Toeplitz(K), cols 64+n -> 2Re(Cb z^{j+1}), 128+n -> -2Im(Cb z^{j+1})
  _Float16* a3 = A3 + (size_t)h*64*192;
  int j = n;
  for (int k = 0; k < 64; ++k) a3[j*192 + k] = (_Float16)((j >= k) ? Ks[j-k] : 0.f);
  for (int m = 0; m < 64; ++m) {
    a3[j*192 +  64 + m] = (_Float16)( 2.f*(cbr[m]*pr[m][j+1] - cbi[m]*pi[m][j+1]));
    a3[j*192 + 128 + m] = (_Float16)(-2.f*(cbr[m]*pi[m][j+1] + cbi[m]*pr[m][j+1]));
  }
  ZT[h*64 + n] = make_float2(pr[n][64], pi[n][64]);
}

// Phase 1: per (chunk,h): S_local(128x32) = A1(128x64) @ U_tile(64x32)   [WMMA f16]
// 4 chunks per workgroup: A panel staged once, reused 4x.
__global__ void s4_phase1(const _Float16* __restrict__ A1, const float* __restrict__ U,
                          float* __restrict__ S) {
  int c0 = blockIdx.x * 4, h = blockIdx.y;
  __shared__ alignas(16) _Float16 As[128*72];   // padded rows: 36 dwords
  __shared__ alignas(16) _Float16 Bs[32*72];    // column-major u panel
  const uint32_t* src = (const uint32_t*)(A1 + (size_t)h*128*64);
  uint32_t* dstA = (uint32_t*)As;
  for (int i = threadIdx.x; i < 4096; i += 256) {
    int row = i >> 5, kd = i & 31;
    dstA[row*36 + kd] = src[i];
  }
  int wave = threadIdx.x >> 5, lane = threadIdx.x & 31;
  int b = threadIdx.x >> 3, kb = (threadIdx.x & 7) * 8;
  for (int cc = 0; cc < 4; ++cc) {
    int c = c0 + cc;
    if (cc) __syncthreads();               // waves done reading previous Bs
    const float* up = U + ((size_t)b*Hn + h)*Ln + c*64 + kb;
    if (cc < 3) __builtin_prefetch(up + 64, 0, 2);   // next chunk's panel
#pragma unroll
    for (int k = 0; k < 8; ++k) Bs[b*72 + kb + k] = (_Float16)up[k];
    __syncthreads();
    float* Sout = S + (((size_t)c*Hn + h)*128)*32;
#pragma unroll
    for (int nt = 0; nt < 2; ++nt) {
      v8f acc = {};
#pragma unroll
      for (int ks = 0; ks < 64; ks += 32) {
        v16h a = frag_a(As, 72, wave*16, ks, lane);
        v16h bb = frag_b(Bs, 72, ks, nt*16, lane);
        acc = __builtin_amdgcn_wmma_f32_16x16x32_f16(false, a, false, bb, (short)0, acc, false, false);
      }
      int col = nt*16 + (lane & 15), r0 = wave*16 + (lane >> 4)*8;
#pragma unroll
      for (int r = 0; r < 8; ++r) Sout[(size_t)(r0+r)*32 + col] = acc[r];
    }
  }
}

// Phase 2: sequential scan over chunks (in place: S_local -> S_in)
__global__ void s4_scan(const float2* __restrict__ ZT, float* __restrict__ S) {
  int idx = blockIdx.x*256 + threadIdx.x;  // B*H*N = 131072
  int b = idx & 31, n = (idx >> 5) & 63, h = idx >> 11;
  float2 zt = ZT[h*64 + n];
  float sre = 0.f, sim = 0.f;
  for (int c = 0; c < NCn; ++c) {
    size_t base = (((size_t)c*Hn + h)*128)*32 + b;
    float lre = S[base + (size_t)n*32];
    float lim = S[base + (size_t)(64+n)*32];
    S[base + (size_t)n*32]      = sre;   // state entering chunk c
    S[base + (size_t)(64+n)*32] = sim;
    float nre = zt.x*sre - zt.y*sim + lre;
    float nim = zt.x*sim + zt.y*sre + lim;
    sre = nre; sim = nim;
  }
}

// Phase 3: y(64x32) = A3(64x192) @ [U_tile; S_in]; then +Dskip*u, gelu -> G (f16)
__global__ void s4_phase3(const _Float16* __restrict__ A3, const float* __restrict__ U,
                          const float* __restrict__ Sin, const float* __restrict__ Dskip,
                          _Float16* __restrict__ G) {
  int c0 = blockIdx.x * 4, h = blockIdx.y;
  __shared__ alignas(16) _Float16 As[64*200];   // padded rows: 100 dwords
  __shared__ alignas(16) _Float16 Bs[32*200];   // column-major [u ; s_in] panel
  const uint32_t* src = (const uint32_t*)(A3 + (size_t)h*64*192);
  uint32_t* dstA = (uint32_t*)As;
  for (int i = threadIdx.x; i < 6144; i += 256) {
    int row = i / 96, kd = i - row*96;
    dstA[row*100 + kd] = src[i];
  }
  int wave = threadIdx.x >> 5, lane = threadIdx.x & 31;
  int mt = wave >> 1, nt = wave & 1;
  int bb_ = threadIdx.x >> 3, kb = (threadIdx.x & 7) * 8;
  float dsk = Dskip[h];
  for (int cc = 0; cc < 4; ++cc) {
    int c = c0 + cc;
    if (cc) __syncthreads();
    {
      const float* up = U + ((size_t)bb_*Hn + h)*Ln + c*64 + kb;
      if (cc < 3) __builtin_prefetch(up + 64, 0, 2);
#pragma unroll
      for (int k = 0; k < 8; ++k) Bs[bb_*200 + kb + k] = (_Float16)up[k];
    }
    {
      const float* sp = Sin + (((size_t)c*Hn + h)*128)*32;
      for (int i = threadIdx.x; i < 128*32; i += 256)
        Bs[(i & 31)*200 + 64 + (i >> 5)] = (_Float16)sp[i];
    }
    __syncthreads();
    v8f acc = {};
#pragma unroll
    for (int ks = 0; ks < 192; ks += 32) {
      v16h a = frag_a(As, 200, mt*16, ks, lane);
      v16h bf = frag_b(Bs, 200, ks, nt*16, lane);
      acc = __builtin_amdgcn_wmma_f32_16x16x32_f16(false, a, false, bf, (short)0, acc, false, false);
    }
    int bcol = nt*16 + (lane & 15);
    int j0 = mt*16 + (lane >> 4)*8;
    const float* up = U + ((size_t)bcol*Hn + h)*Ln + c*64;
    _Float16* gp = G + ((size_t)bcol*Hn + h)*Ln + c*64;
#pragma unroll
    for (int r = 0; r < 8; ++r) {
      int j = j0 + r;
      gp[j] = (_Float16)gelu_tanh(acc[r] + dsk*up[j]);
    }
  }
}

// Output projection: out(64 x 64L) = Wo(64x64) @ G + bo + u   [WMMA f16]
// 4 l-tiles per workgroup; G transposed at LDS-write time (global reads coalesced).
__global__ void s4_wo(const float* __restrict__ Wo, const float* __restrict__ bo,
                      const _Float16* __restrict__ G, const float* __restrict__ U,
                      float* __restrict__ Out) {
  int lt0 = blockIdx.x * 4, b = blockIdx.y;
  __shared__ alignas(16) _Float16 As[64*72];    // Wo, padded rows
  __shared__ alignas(16) _Float16 Bs[64*72];    // G tile, column-major by l
  for (int i = threadIdx.x; i < 4096; i += 256) {
    int row = i >> 6, k = i & 63;
    As[row*72 + k] = (_Float16)Wo[i];
  }
  int wave = threadIdx.x >> 5, lane = threadIdx.x & 31;
  int hh = threadIdx.x >> 2, lb = (threadIdx.x & 3) * 16;
  for (int tt = 0; tt < 4; ++tt) {
    int lt = lt0 + tt;
    if (tt) __syncthreads();
    const _Float16* gp = G + ((size_t)b*Hn + hh)*Ln + lt*64 + lb;
    if (tt < 3) __builtin_prefetch(gp + 64, 0, 2);
#pragma unroll
    for (int l = 0; l < 16; ++l) Bs[(lb + l)*72 + hh] = gp[l];
    __syncthreads();
#pragma unroll
    for (int t = wave; t < 16; t += 8) {
      int mt = t >> 2, nt = t & 3;
      v8f acc = {};
#pragma unroll
      for (int ks = 0; ks < 64; ks += 32) {
        v16h a = frag_a(As, 72, mt*16, ks, lane);
        v16h bf = frag_b(Bs, 72, ks, nt*16, lane);
        acc = __builtin_amdgcn_wmma_f32_16x16x32_f16(false, a, false, bf, (short)0, acc, false, false);
      }
      int l = lt*64 + nt*16 + (lane & 15);
      int o0 = mt*16 + (lane >> 4)*8;
#pragma unroll
      for (int r = 0; r < 8; ++r) {
        int o = o0 + r;
        size_t off = ((size_t)b*Hn + o)*Ln + l;
        Out[off] = acc[r] + bo[o] + U[off];
      }
    }
  }
}

// ===========================================================================
// Conv / BN kernels (bandwidth-bound, VALU)
// ===========================================================================

__global__ void conv_in(const float* __restrict__ X, const float* __restrict__ W,
                        const float* __restrict__ Bias, float* __restrict__ Y) {
  int b = blockIdx.y, l0 = blockIdx.x * 256, t = threadIdx.x;
  __shared__ float xs[258];
  const float* xp = X + (size_t)b*Ln;
  for (int i = t; i < 258; i += 256) {
    int l = l0 - 1 + i;
    xs[i] = (l >= 0 && l < Ln) ? xp[l] : 0.f;
  }
  __syncthreads();
  float x0 = xs[t], x1 = xs[t+1], x2 = xs[t+2];
  for (int o = 0; o < 64; ++o)
    Y[((size_t)b*Hn + o)*Ln + l0 + t] = W[o*3]*x0 + W[o*3+1]*x1 + W[o*3+2]*x2 + Bias[o];
}

__global__ void conv_mid(const float* __restrict__ X, const float* __restrict__ W,
                         const float* __restrict__ Bias, float* __restrict__ Y) {
  int b = blockIdx.y, l0 = blockIdx.x * 64;
  __shared__ float xs[64][66];
  __shared__ float ws[64*64*3];
  for (int i = threadIdx.x; i < 64*66; i += 256) {
    int c = i / 66, l = i % 66, gl = l0 - 1 + l;
    xs[c][l] = (gl >= 0 && gl < Ln) ? X[((size_t)b*Hn + c)*Ln + gl] : 0.f;
  }
  for (int i = threadIdx.x; i < 64*64*3; i += 256) ws[i] = W[i];
  __syncthreads();
  int o = threadIdx.x & 63, lg = threadIdx.x >> 6;
  for (int li = 0; li < 16; ++li) {
    int l = lg*16 + li;
    float acc = Bias[o];
    for (int c = 0; c < 64; ++c) {
      const float* wp = &ws[(o*64 + c)*3];
      acc += wp[0]*xs[c][l] + wp[1]*xs[c][l+1] + wp[2]*xs[c][l+2];
    }
    Y[((size_t)b*Hn + o)*Ln + l0 + l] = acc;
  }
}

__global__ void conv_last(const float* __restrict__ X, const float* __restrict__ W,
                          const float* __restrict__ Bias, float* __restrict__ Y) {
  int b = blockIdx.y, l0 = blockIdx.x * 256, t = threadIdx.x;
  __shared__ float xs[64][258];
  for (int i = threadIdx.x; i < 64*258; i += 256) {
    int c = i / 258, l = i % 258, gl = l0 - 1 + l;
    xs[c][l] = (gl >= 0 && gl < Ln) ? X[((size_t)b*Hn + c)*Ln + gl] : 0.f;
  }
  __syncthreads();
  float acc = Bias[0];
  for (int c = 0; c < 64; ++c)
    acc += W[c*3]*xs[c][t] + W[c*3+1]*xs[c][t+1] + W[c*3+2]*xs[c][t+2];
  Y[(size_t)b*Ln + l0 + t] = acc;
}

// Deterministic two-stage BN stats (no atomics -> graph replays bit-stable)
__global__ void bn_stats(const float* __restrict__ Y, float* __restrict__ P) {
  int ch = blockIdx.x, b = blockIdx.y;
  const float* yp = Y + ((size_t)b*Hn + ch)*Ln;
  float s = 0.f, s2 = 0.f;
  for (int i = threadIdx.x; i < Ln; i += 256) { float v = yp[i]; s += v; s2 += v*v; }
  __shared__ float sh[256], sh2[256];
  sh[threadIdx.x] = s; sh2[threadIdx.x] = s2;
  __syncthreads();
  for (int off = 128; off > 0; off >>= 1) {
    if (threadIdx.x < off) { sh[threadIdx.x] += sh[threadIdx.x+off]; sh2[threadIdx.x] += sh2[threadIdx.x+off]; }
    __syncthreads();
  }
  if (threadIdx.x == 0) { P[ch*32 + b] = sh[0]; P[2048 + ch*32 + b] = sh2[0]; }
}

__global__ void bn_reduce(const float* __restrict__ P, float* __restrict__ F) {
  int ch = threadIdx.x;
  float s = 0.f, s2 = 0.f;
  for (int b = 0; b < 32; ++b) { s += P[ch*32 + b]; s2 += P[2048 + ch*32 + b]; }
  const float cnt = (float)Bn * (float)Ln;
  float m = s / cnt;
  float v = s2 / cnt - m*m;
  F[ch] = m;
  F[64 + ch] = rsqrtf(v + 1e-5f);
}

__global__ void bn_relu(const float* __restrict__ F, const float* __restrict__ g,
                        const float* __restrict__ be, float* __restrict__ Y) {
  size_t idx = (size_t)blockIdx.x*256 + threadIdx.x;
  int ch = (int)((idx >> 12) & 63);
  float y = (Y[idx] - F[ch]) * F[64 + ch] * g[ch] + be[ch];
  Y[idx] = y > 0.f ? y : 0.f;
}

// ===========================================================================
extern "C" void kernel_launch(void* const* d_in, const int* in_sizes, int n_in,
                              void* d_out, int out_size, void* d_ws, size_t ws_size,
                              hipStream_t stream) {
  if (ws_size < WS_NEED) return;
  const float* x    = (const float*)d_in[0];
  const float* w1   = (const float*)d_in[1];  const float* b1  = (const float*)d_in[2];
  const float* g1   = (const float*)d_in[3];  const float* be1 = (const float*)d_in[4];
  const float* w9   = (const float*)d_in[5];  const float* b9  = (const float*)d_in[6];
  const float* g9   = (const float*)d_in[7];  const float* be9 = (const float*)d_in[8];
  const float* w16  = (const float*)d_in[9];  const float* b16 = (const float*)d_in[10];
  const float* g16  = (const float*)d_in[11]; const float* be16= (const float*)d_in[12];
  const float* w17  = (const float*)d_in[13]; const float* b17 = (const float*)d_in[14];
  const float* log_dt   = (const float*)d_in[15];
  const float* A_re_log = (const float*)d_in[16];
  const float* A_im     = (const float*)d_in[17];
  const float* C_re     = (const float*)d_in[18];
  const float* C_im     = (const float*)d_in[19];
  const float* Dskip    = (const float*)d_in[20];
  const float* Wo       = (const float*)d_in[21];
  const float* bo       = (const float*)d_in[22];

  char* ws = (char*)d_ws;
  float*     bufA  = (float*)(ws + OFF_BUFA);
  float*     bufB  = (float*)(ws + OFF_BUFB);
  float*     S     = (float*)(ws + OFF_S);
  _Float16*  G     = (_Float16*)(ws + OFF_G);
  _Float16*  A1    = (_Float16*)(ws + OFF_A1);
  _Float16*  A3    = (_Float16*)(ws + OFF_A3);
  float2*    ZT    = (float2*)(ws + OFF_ZT);
  float4*    zcb   = (float4*)(ws + OFF_ZCB);
  float*     statP = (float*)(ws + OFF_STP);
  float*     statF = (float*)(ws + OFF_STF);

  float* cur = bufA; float* nxt = bufB;

  auto bn = [&](float* buf, const float* g, const float* be) {
    bn_stats<<<dim3(64,32),256,0,stream>>>(buf, statP);
    bn_reduce<<<1,64,0,stream>>>(statP, statF);
    bn_relu<<<32768,256,0,stream>>>(statF, g, be, buf);
  };
  auto s4block = [&](int i) {
    s4_prep    <<<16,256,0,stream>>>(log_dt+i*Hn, A_re_log+(size_t)i*Hn*Nn, A_im+(size_t)i*Hn*Nn,
                                     C_re+(size_t)i*Hn*Nn, C_im+(size_t)i*Hn*Nn, zcb);
    s4_chunkmat<<<64,64,0,stream>>>(zcb, A1, A3, ZT);
    s4_phase1  <<<dim3(16,Hn),256,0,stream>>>(A1, cur, S);
    s4_scan    <<<512,256,0,stream>>>(ZT, S);
    s4_phase3  <<<dim3(16,Hn),256,0,stream>>>(A3, cur, S, Dskip+i*Hn, G);
    s4_wo      <<<dim3(16,32),256,0,stream>>>(Wo+(size_t)i*Hn*Hn, bo+i*Hn, G, cur, nxt);
    float* t = cur; cur = nxt; nxt = t;
  };

  // layer 1
  conv_in<<<dim3(16,32),256,0,stream>>>(x, w1, b1, cur);
  bn(cur, g1, be1);
  // layers 2-8
  for (int i = 0; i < 7; ++i) s4block(i);
  // layer 9
  conv_mid<<<dim3(64,32),256,0,stream>>>(cur, w9, b9, nxt);
  { float* t = cur; cur = nxt; nxt = t; }
  bn(cur, g9, be9);
  // layers 10-15
  for (int i = 7; i < 13; ++i) s4block(i);
  // layer 16
  conv_mid<<<dim3(64,32),256,0,stream>>>(cur, w16, b16, nxt);
  { float* t = cur; cur = nxt; nxt = t; }
  bn(cur, g16, be16);
  // layer 17
  conv_last<<<dim3(16,32),256,0,stream>>>(cur, w17, b17, (float*)d_out);
}
